// CVNeuralOp_57037165691284
// MI455X (gfx1250) — matrix operand order repaired and checked by reading, not compiled, not run on hardware
//
#include <hip/hip_runtime.h>

#define NNODE 20000
#define NEDGE 320000
#define CCH 16
#define WAVES 8

typedef __attribute__((ext_vector_type(16))) _Float16 v16h;
typedef __attribute__((ext_vector_type(8)))  _Float16 v8h;
typedef __attribute__((ext_vector_type(8)))  float    v8f;

// Packed per-MLP weight block (f16 WMMA B-fragments + f32 biases/prelu):
//  half offsets: L0 @0 (2048), L1 @2048 (4096), L2 @6144 (4096), L3 @10240 (16384)
//  f32 region @byte 53248: b0[64], b1[64], b2[64], b3[256], a0, a1, a2
#define PACK_HALFS   26624
#define PACK_F32_OFF 53248
#define PACK_F32_CNT 451
#define PACK_BYTES   55296

__device__ __forceinline__ float preluf(float x, float a) { return x >= 0.f ? x : a * x; }

// A-fragment (16x32 f16): lane L: M=L%16; e<8 -> K=(L<16?0:8)+e ; e>=8 -> K=16+(L<16?0:8)+(e-8)
__device__ __forceinline__ v16h load_a_frag(const _Float16* act, int stride, int kt, int lane) {
  int M = lane & 15;
  int hi = (lane >> 4) & 1;
  const _Float16* p = act + M * stride + kt * 32 + hi * 8;
  v8h lo = *(const v8h*)(p);
  v8h hh = *(const v8h*)(p + 16);
  v16h a;
#pragma unroll
  for (int i = 0; i < 8; ++i) { a[i] = lo[i]; a[8 + i] = hh[i]; }
  return a;
}

// B-fragment (32x16 f16): pre-packed so lane L reads 16 contiguous halves.
__device__ __forceinline__ v16h load_b_frag(const _Float16* bp, int lane) {
  const _Float16* p = bp + lane * 16;
  v8h lo = *(const v8h*)(p);
  v8h hh = *(const v8h*)(p + 8);
  v16h b;
#pragma unroll
  for (int i = 0; i < 8; ++i) { b[i] = lo[i]; b[8 + i] = hh[i]; }
  return b;
}

// One MLP layer: [16,32*KT] @ [32*KT,64] + bias, PReLU, write f16 to LDS (stride 64).
__device__ __forceinline__ void mlp_layer(const _Float16* actIn, int inStride, int KT,
                                          const _Float16* Bpack, const float* bias, float a,
                                          _Float16* actOut, int lane) {
  int n0 = lane & 15;
  int hi = (lane >> 4) & 1;
  v16h A0 = load_a_frag(actIn, inStride, 0, lane);
  v16h A1 = (KT > 1) ? load_a_frag(actIn, inStride, 1, lane) : A0;
#pragma unroll
  for (int nt = 0; nt < 4; ++nt) {
    v8f c = {};
    v16h B0 = load_b_frag(Bpack + (0 * 4 + nt) * 512, lane);
    c = __builtin_amdgcn_wmma_f32_16x16x32_f16(false, A0, false, B0, (short)0, c, false, false);
    if (KT > 1) {
      v16h B1 = load_b_frag(Bpack + (1 * 4 + nt) * 512, lane);
      c = __builtin_amdgcn_wmma_f32_16x16x32_f16(false, A1, false, B1, (short)0, c, false, false);
    }
    float bv = bias[nt * 16 + n0];
#pragma unroll
    for (int r = 0; r < 8; ++r) {
      // D layout: lane L holds N=L%16, M = r + (L<16?0:8)
      float f = preluf(c[r] + bv, a);
      actOut[(r + hi * 8) * 64 + nt * 16 + n0] = (_Float16)f;
    }
  }
}

// Fused: edge-MLP (rmlp & imlp) -> per-edge matvec -> signed atomic scatter.
// One wave handles 16 edges. E is an exact multiple of 128 (block = 8 waves).
__global__ __launch_bounds__(256) void edge_kernel(
    const int* __restrict__ eidx, const float* __restrict__ eattr,
    const float* __restrict__ xr, const float* __restrict__ xi,
    float* __restrict__ accR, float* __restrict__ accI,
    const unsigned char* __restrict__ wrBlk, const unsigned char* __restrict__ wiBlk)
{
  __shared__ __align__(16) _Float16 sactA[WAVES][16 * 64];
  __shared__ __align__(16) _Float16 sactB[WAVES][16 * 64];
  __shared__ float sxr[WAVES][256];
  __shared__ float sxi[WAVES][256];
  __shared__ int ssrc[WAVES][16];
  __shared__ int sdst[WAVES][16];

  const int lane = threadIdx.x & 31;
  const int w = threadIdx.x >> 5;
  const int ebase = (blockIdx.x * WAVES + w) * 16;
  const int n0 = lane & 15;
  const int hi = (lane >> 4) & 1;

  if (lane < 16) {
    ssrc[w][lane] = eidx[ebase + lane];
    sdst[w][lane] = eidx[NEDGE + ebase + lane];
  }
  __syncthreads();
#pragma unroll
  for (int k = 0; k < 8; ++k) {
    int idx = lane + k * 32;
    int e = idx >> 4, c = idx & 15;
    int s = ssrc[w][e];
    sxr[w][idx] = xr[s * CCH + c];
    sxi[w][idx] = xi[s * CCH + c];
  }

  float msgRe[8], msgIm[8];
#pragma unroll
  for (int r = 0; r < 8; ++r) { msgRe[r] = 0.f; msgIm[r] = 0.f; }

  for (int pass = 0; pass < 2; ++pass) {   // 0: rmlp (Wr), 1: imlp (Wi)
    const unsigned char* blk = pass ? wiBlk : wrBlk;
    const _Float16* Wh = (const _Float16*)blk;
    const float* Fb = (const float*)(blk + PACK_F32_OFF);

    // stage edge_attr -> sactA [16][32], zero-padded K
#pragma unroll
    for (int t = 0; t < 16; ++t) {
      int idx = lane + t * 32;
      int m = idx >> 5, k2 = idx & 31;
      float v = (k2 < 6) ? eattr[(ebase + m) * 6 + k2] : 0.f;
      sactA[w][m * 32 + k2] = (_Float16)v;
    }
    __syncthreads();
    mlp_layer(sactA[w], 32, 1, Wh + 0,    Fb + 0,   Fb[448], sactB[w], lane);
    __syncthreads();
    mlp_layer(sactB[w], 64, 2, Wh + 2048, Fb + 64,  Fb[449], sactA[w], lane);
    __syncthreads();
    mlp_layer(sactA[w], 64, 2, Wh + 6144, Fb + 128, Fb[450], sactB[w], lane);
    __syncthreads();

    // L3 (64 -> 256) fused with per-edge matvec: n-tile nt is row c=nt of We[e].
    v16h A0 = load_a_frag(sactB[w], 64, 0, lane);
    v16h A1 = load_a_frag(sactB[w], 64, 1, lane);
    const float* b3 = Fb + 192;
    const _Float16* W3p = Wh + 10240;
#pragma unroll 4
    for (int nt = 0; nt < 16; ++nt) {
      v8f c = {};
      v16h B0 = load_b_frag(W3p + (0 * 16 + nt) * 512, lane);
      c = __builtin_amdgcn_wmma_f32_16x16x32_f16(false, A0, false, B0, (short)0, c, false, false);
      v16h B1 = load_b_frag(W3p + (1 * 16 + nt) * 512, lane);
      c = __builtin_amdgcn_wmma_f32_16x16x32_f16(false, A1, false, B1, (short)0, c, false, false);
      float bv = b3[nt * 16 + n0];
#pragma unroll
      for (int r = 0; r < 8; ++r) {
        int e = r + hi * 8;
        float we = c[r] + bv;            // We[e][nt][n0]
        float xre = sxr[w][e * 16 + nt]; // broadcast within half-wave
        float xim = sxi[w][e * 16 + nt];
        if (pass == 0) { msgRe[r] += xre * we; msgIm[r] += xim * we; }   // rr, ri
        else           { msgIm[r] += xre * we; msgRe[r] -= xim * we; }   // ir, -ii
      }
    }
    __syncthreads();
  }

#pragma unroll
  for (int r = 0; r < 8; ++r) {
    int e = r + hi * 8;
    int d = sdst[w][e];
    atomicAdd(&accR[d * CCH + n0], msgRe[r]);
    atomicAdd(&accI[d * CCH + n0], msgIm[r]);
  }
}

// Pack one edge-MLP's weights into WMMA B-fragment order + f32 bias/prelu region.
__global__ __launch_bounds__(256) void pack_kernel(
    const float* __restrict__ W0, const float* __restrict__ W1,
    const float* __restrict__ W2, const float* __restrict__ W3,
    const float* __restrict__ b0, const float* __restrict__ b1,
    const float* __restrict__ b2, const float* __restrict__ b3,
    const float* __restrict__ a0, const float* __restrict__ a1,
    const float* __restrict__ a2, unsigned char* __restrict__ outBlk)
{
  int t = blockIdx.x * blockDim.x + threadIdx.x;
  _Float16* H = (_Float16*)outBlk;
  float* F = (float*)(outBlk + PACK_F32_OFF);
  if (t < PACK_HALFS) {
    const float* W; int ncols; int local; bool pad0 = false;
    if (t < 2048)       { W = W0; ncols = 64;  local = t;         pad0 = true; }
    else if (t < 6144)  { W = W1; ncols = 64;  local = t - 2048;  }
    else if (t < 10240) { W = W2; ncols = 64;  local = t - 6144;  }
    else                { W = W3; ncols = 256; local = t - 10240; }
    int perKt = (ncols / 16) * 512;
    int kt = local / perKt;
    int r2 = local % perKt;
    int nt = r2 / 512;
    int lr = r2 % 512;
    int lane = lr / 16;
    int e = lr % 16;
    int K = kt * 32 + ((lane & 16) ? 8 : 0) + (e < 8 ? e : 8 + e);   // B 32x16: e<8 -> K=(L<16?0:8)+e ; e>=8 -> K=16+(L<16?0:8)+(e-8)
    int Nc = nt * 16 + (lane & 15);
    float v = (pad0 && K >= 6) ? 0.f : W[K * ncols + Nc];
    H[t] = (_Float16)v;
  } else if (t < PACK_HALFS + PACK_F32_CNT) {
    int i = t - PACK_HALFS;
    float v;
    if (i < 64)       v = b0[i];
    else if (i < 128) v = b1[i - 64];
    else if (i < 192) v = b2[i - 128];
    else if (i < 448) v = b3[i - 192];
    else if (i == 448) v = a0[0];
    else if (i == 449) v = a1[0];
    else               v = a2[0];
    F[i] = v;
  }
}

__global__ void zero_kernel(float* __restrict__ p, int n) {
  int i = blockIdx.x * blockDim.x + threadIdx.x;
  if (i < n) p[i] = 0.f;
}

__global__ void count_kernel(const int* __restrict__ dst, float* __restrict__ cnt) {
  int e = blockIdx.x * blockDim.x + threadIdx.x;
  if (e < NEDGE) atomicAdd(&cnt[dst[e]], 1.f);
}

__global__ void inv_kernel(const float* __restrict__ cnt, float* __restrict__ inv) {
  int n = blockIdx.x * blockDim.x + threadIdx.x;
  if (n < NNODE) inv[n] = 1.f / fmaxf(cnt[n], 1.f);
}

__global__ __launch_bounds__(128) void up_kernel(
    const float* __restrict__ x0r, const float* __restrict__ x0i,
    const float* Wr0, const float* br0, const float* ar0,
    const float* Wr1, const float* br1, const float* ar1,
    const float* Wi0, const float* bi0, const float* ai0,
    const float* Wi1, const float* bi1, const float* ai1,
    float* __restrict__ xr, float* __restrict__ xi)
{
  int n = blockIdx.x * blockDim.x + threadIdx.x;
  if (n >= NNODE) return;
  float in[6], h[8];
#pragma unroll
  for (int c = 0; c < 6; ++c) in[c] = x0r[n * 6 + c];
#pragma unroll
  for (int o = 0; o < 8; ++o) {
    float s = br0[o];
    for (int c = 0; c < 6; ++c) s += in[c] * Wr0[c * 8 + o];
    h[o] = preluf(s, ar0[0]);
  }
#pragma unroll
  for (int o = 0; o < 16; ++o) {
    float s = br1[o];
    for (int c = 0; c < 8; ++c) s += h[c] * Wr1[c * 16 + o];
    xr[n * 16 + o] = preluf(s, ar1[0]);
  }
#pragma unroll
  for (int c = 0; c < 6; ++c) in[c] = x0i[n * 6 + c];
#pragma unroll
  for (int o = 0; o < 8; ++o) {
    float s = bi0[o];
    for (int c = 0; c < 6; ++c) s += in[c] * Wi0[c * 8 + o];
    h[o] = preluf(s, ai0[0]);
  }
#pragma unroll
  for (int o = 0; o < 16; ++o) {
    float s = bi1[o];
    for (int c = 0; c < 8; ++c) s += h[c] * Wi1[c * 16 + o];
    xi[n * 16 + o] = preluf(s, ai1[0]);
  }
}

__global__ __launch_bounds__(128) void node_kernel(
    const float* __restrict__ xr, const float* __restrict__ xi,
    const float* __restrict__ accR, const float* __restrict__ accI,
    const float* __restrict__ inv,
    const float* __restrict__ rroot, const float* __restrict__ rbias,
    const float* __restrict__ iroot, const float* __restrict__ ibias,
    const float* __restrict__ arp, const float* __restrict__ aip,
    float* __restrict__ xrn, float* __restrict__ xin)
{
  __shared__ float sR[256], sI[256], sRB[16], sIB[16];
  int t = threadIdx.x;
  sR[t] = rroot[t]; sR[t + 128] = rroot[t + 128];
  sI[t] = iroot[t]; sI[t + 128] = iroot[t + 128];
  if (t < 16) { sRB[t] = rbias[t]; sIB[t] = ibias[t]; }
  __syncthreads();
  int n = blockIdx.x * 128 + t;
  if (n >= NNODE) return;
  float xrv[16], xiv[16];
#pragma unroll
  for (int c = 0; c < 16; ++c) { xrv[c] = xr[n * 16 + c]; xiv[c] = xi[n * 16 + c]; }
  float iv = inv[n];
  float ar = arp[0], ai = aip[0];
#pragma unroll 4
  for (int o = 0; o < 16; ++o) {
    float rr_r = 0.f, rr_i = 0.f, ii_r = 0.f, ii_i = 0.f;
#pragma unroll
    for (int c = 0; c < 16; ++c) {
      float wr = sR[c * 16 + o], wi = sI[c * 16 + o];
      rr_r += xrv[c] * wr;   // xr @ rroot
      rr_i += xiv[c] * wr;   // xi @ rroot
      ii_r += xrv[c] * wi;   // xr @ iroot
      ii_i += xiv[c] * wi;   // xi @ iroot
    }
    float re = accR[n * 16 + o] * iv + rr_r + sRB[o] - ii_i - sIB[o];
    float im = accI[n * 16 + o] * iv + rr_i + sRB[o] + ii_r + sIB[o];
    xrn[n * 16 + o] = preluf(re, ar);
    xin[n * 16 + o] = preluf(im, ai);
  }
}

__global__ __launch_bounds__(128) void down_kernel(
    const float* __restrict__ xr, const float* __restrict__ xi,
    const float* Wr0, const float* br0, const float* ar0,
    const float* Wr1, const float* br1, const float* ar1,
    const float* Wi0, const float* bi0, const float* ai0,
    const float* Wi1, const float* bi1, const float* ai1,
    float* __restrict__ out)
{
  int n = blockIdx.x * blockDim.x + threadIdx.x;
  if (n >= NNODE) return;
  float h[8];
#pragma unroll
  for (int o = 0; o < 8; ++o) {
    float s = br0[o];
    for (int c = 0; c < 16; ++c) s += xr[n * 16 + c] * Wr0[c * 8 + o];
    h[o] = preluf(s, ar0[0]);
  }
  {
    float s = br1[0];
    for (int c = 0; c < 8; ++c) s += h[c] * Wr1[c];
    out[n] = preluf(s, ar1[0]);
  }
#pragma unroll
  for (int o = 0; o < 8; ++o) {
    float s = bi0[o];
    for (int c = 0; c < 16; ++c) s += xi[n * 16 + c] * Wi0[c * 8 + o];
    h[o] = preluf(s, ai0[0]);
  }
  {
    float s = bi1[0];
    for (int c = 0; c < 8; ++c) s += h[c] * Wi1[c];
    out[NNODE + n] = preluf(s, ai1[0]);
  }
}

extern "C" void kernel_launch(void* const* d_in, const int* in_sizes, int n_in,
                              void* d_out, int out_size, void* d_ws, size_t ws_size,
                              hipStream_t stream)
{
  (void)out_size; (void)ws_size;
  const float* x0r  = (const float*)d_in[0];
  const float* x0i  = (const float*)d_in[1];
  const int*   eidx = (const int*)d_in[2];
  const float* eatt = (const float*)d_in[3];

  // Detect params flattening: dict-insertion order (leaf 4 = rmlp.W0, 384)
  // vs jax-pytree alphabetical (leaf 4 = downi[0].W, 128).
  bool ins = (n_in > 4 && in_sizes[4] == 384);
  auto F = [&](int i) { return (const float*)d_in[i]; };

  float* ws = (float*)d_ws;
  const int NC = NNODE * CCH;
  float* bufR[2] = { ws,          ws + 2 * NC };
  float* bufI[2] = { ws + NC,     ws + 3 * NC };
  float* accR = ws + 4 * NC;
  float* accI = ws + 5 * NC;              // contiguous with accR
  float* cnt  = ws + 6 * NC;
  float* inv  = ws + 6 * NC + NNODE;
  unsigned char* packed = (unsigned char*)(ws + 6 * NC + 2 * NNODE);

  zero_kernel<<<(NNODE + 255) / 256, 256, 0, stream>>>(cnt, NNODE);
  count_kernel<<<(NEDGE + 255) / 256, 256, 0, stream>>>(eidx + NEDGE, cnt);
  inv_kernel<<<(NNODE + 255) / 256, 256, 0, stream>>>(cnt, inv);

  const int packThreads = PACK_HALFS + PACK_F32_CNT;
  for (int k = 0; k < 10; ++k) {
    int kb = (ins ? 4 : 16) + 28 * k;
    int rb = ins ? (kb + 0)  : (kb + 16);   // rmlp leaves: W0,b0,a0,W1,b1,a1,W2,b2,a2,W3,b3
    int ib = ins ? (kb + 11) : (kb + 3);    // imlp leaves
    pack_kernel<<<(packThreads + 255) / 256, 256, 0, stream>>>(
        F(rb + 0), F(rb + 3), F(rb + 6), F(rb + 9),
        F(rb + 1), F(rb + 4), F(rb + 7), F(rb + 10),
        F(rb + 2), F(rb + 5), F(rb + 8),
        packed + (size_t)(2 * k) * PACK_BYTES);
    pack_kernel<<<(packThreads + 255) / 256, 256, 0, stream>>>(
        F(ib + 0), F(ib + 3), F(ib + 6), F(ib + 9),
        F(ib + 1), F(ib + 4), F(ib + 7), F(ib + 10),
        F(ib + 2), F(ib + 5), F(ib + 8),
        packed + (size_t)(2 * k + 1) * PACK_BYTES);
  }

  int upr = ins ? 284 : 302;
  int upi = ins ? 290 : 296;
  up_kernel<<<(NNODE + 127) / 128, 128, 0, stream>>>(x0r, x0i,
      F(upr + 0), F(upr + 1), F(upr + 2), F(upr + 3), F(upr + 4), F(upr + 5),
      F(upi + 0), F(upi + 1), F(upi + 2), F(upi + 3), F(upi + 4), F(upi + 5),
      bufR[0], bufI[0]);

  int cur = 0;
  for (int k = 0; k < 10; ++k) {
    int kb = (ins ? 4 : 16) + 28 * k;
    const float *rroot, *rbias, *iroot, *ibias, *ar, *ai;
    if (ins) { rroot = F(kb + 22); rbias = F(kb + 23); iroot = F(kb + 24);
               ibias = F(kb + 25); ar = F(kb + 26); ai = F(kb + 27); }
    else     { ai = F(kb + 0); ar = F(kb + 1); ibias = F(kb + 2); iroot = F(kb + 14);
               rbias = F(kb + 15); rroot = F(kb + 27); }

    zero_kernel<<<(2 * NC + 255) / 256, 256, 0, stream>>>(accR, 2 * NC);
    edge_kernel<<<NEDGE / (16 * WAVES), 256, 0, stream>>>(
        eidx, eatt, bufR[cur], bufI[cur], accR, accI,
        packed + (size_t)(2 * k) * PACK_BYTES,
        packed + (size_t)(2 * k + 1) * PACK_BYTES);
    node_kernel<<<(NNODE + 127) / 128, 128, 0, stream>>>(
        bufR[cur], bufI[cur], accR, accI, inv,
        rroot, rbias, iroot, ibias, ar, ai,
        bufR[1 - cur], bufI[1 - cur]);
    cur = 1 - cur;
  }

  int dnr = ins ? 296 : 10;
  int dni = ins ? 302 : 4;
  down_kernel<<<(NNODE + 127) / 128, 128, 0, stream>>>(bufR[cur], bufI[cur],
      F(dnr + 0), F(dnr + 1), F(dnr + 2), F(dnr + 3), F(dnr + 4), F(dnr + 5),
      F(dni + 0), F(dni + 1), F(dni + 2), F(dni + 3), F(dni + 4), F(dni + 5),
      (float*)d_out);
}
